// hsic_loss_88218628260513
// MI455X (gfx1250) — compile-verified, matrix operand order checked by reading, not run on hardware
//
#include <hip/hip_runtime.h>
#include <hip/hip_bf16.h>
#include <stdint.h>

// ---------------------------------------------------------------------------
// HSIC loss pipeline for MI455X (gfx1250), wave32 + WMMA + TDM.
//   x: [2048, 784] f32, y: [2048, 10] f32  ->  scalar f32
// Stages:
//  1) bf16-convert x (pad K 784->800) + f32 row norms
//  2) D_x = r_i + r_j - 2 * (X X^T): cooperative 32x64 macro-tiles,
//     panels staged in LDS by the Tensor Data Mover (tensor_load_to_lds,
//     double-buffered, s_wait_tensorcnt), math via v_wmma_f32_16x16x32_bf16
//  3) D_y directly from y
//  4) 12 sigmas via 2-level histogram median selection
//  5) HSIC reductions using the centering expansion:
//     mean(Kc_a * Kc_b^T) * m^2 = G_ab - (2/m) dot(S_a,S_b) + sum(S_a)sum(S_b)/m^2
// ---------------------------------------------------------------------------

typedef __bf16 bf16;
typedef __attribute__((ext_vector_type(16))) __bf16 v16bf;
typedef __attribute__((ext_vector_type(8)))  __bf16 v8bf;
typedef __attribute__((ext_vector_type(8)))  float  v8f;
typedef __attribute__((ext_vector_type(4)))  unsigned v4u;
typedef __attribute__((ext_vector_type(8)))  int      v8i;
typedef __attribute__((ext_vector_type(4)))  int      v4i;

#define M_PTS 2048
#define KDIM  784
#define KPAD  800
#define KC    160           // K-chunk staged per TDM tile
#define NCHUNK (KPAD / KC)  // 5
#define DLAT  10
#define NPAIR 65            // pairs a<=b for d=10
#define NBF   4096          // hist bins for Dx/Dy
#define NBC   1024          // hist bins per latent component
#define NSEL  12
#define HSTRIDE 4096
// N-rank in full-matrix counting equivalent to the reference median position
// (multiset weight = 2*fullcount - m; rank = (k/2 + m)/2 with k = 2m^2 - m)
#define RANKSEL 2097664ULL

struct Sel { float lo; float hi; unsigned long long rank; };

// ---- workspace layout (bytes, 256-aligned) --------------------------------
#define OFF_XBF   ((size_t)0)                         // 2048*800*2 = 3,276,800
#define OFF_RX    ((size_t)3276800)                   // 2048*4
#define OFF_DX    ((size_t)3284992)                   // 2048^2*4
#define OFF_DY    ((size_t)20062208)                  // 2048^2*4
#define OFF_HIST  ((size_t)36839424)                  // 12*4096*4
#define OFF_SEL   ((size_t)37036032)                  // 12*16
#define OFF_INV   ((size_t)37036288)                  // 12*4
#define OFF_MAXB  ((size_t)37036544)                  // 2*4
#define OFF_YMM   ((size_t)37036800)                  // 20*4
#define OFF_SCAL  ((size_t)37037056)                  // 16*4
#define OFF_GAB   ((size_t)37037312)                  // 65*4
#define OFF_RSX   ((size_t)37037824)                  // 2048*4
#define OFF_RSY   ((size_t)37046016)                  // 2048*4
#define OFF_RSC   ((size_t)37054208)                  // 10*2048*4 -> end 37,136,128

// ---- helpers --------------------------------------------------------------
__device__ __forceinline__ float block_reduce(float v, float* sm) {
  const int lane = threadIdx.x & 31;
  const int wv   = threadIdx.x >> 5;
#pragma unroll
  for (int o = 16; o > 0; o >>= 1) v += __shfl_down(v, o, 32);
  if (lane == 0) sm[wv] = v;
  __syncthreads();
  const int nw = (int)((blockDim.x + 31) >> 5);
  float t = ((int)threadIdx.x < nw) ? sm[threadIdx.x] : 0.0f;
  if (wv == 0) {
#pragma unroll
    for (int o = 16; o > 0; o >>= 1) t += __shfl_down(t, o, 32);
    if (lane == 0) sm[0] = t;
  }
  __syncthreads();
  float r = sm[0];
  __syncthreads();
  return r;
}

// ---- 1) convert x to bf16 (zero-padded K) + f32 row norms -----------------
__global__ void k_prep_x(const float* __restrict__ x, bf16* __restrict__ xbf,
                         float* __restrict__ rx) {
  __shared__ float sm[32];
  const int row = blockIdx.x;
  const float* xr = x + (size_t)row * KDIM;
  bf16* br = xbf + (size_t)row * KPAD;
  float acc = 0.0f;
  for (int c = threadIdx.x; c < KPAD; c += blockDim.x) {
    float f = (c < KDIM) ? xr[c] : 0.0f;
    br[c] = (bf16)f;
    acc += f * f;
  }
  float t = block_reduce(acc, sm);
  if (threadIdx.x == 0) rx[row] = t;
}

// ---- 2) Gram via TDM-staged LDS panels + WMMA -----------------------------
// TDM D# (2D tile): 16 rows x KC cols of bf16 from row-major [2048 x KPAD]
// tensor into a contiguous LDS tile (row-major, rows of KC elements).
__device__ __forceinline__ void tdm_load_tile(const bf16* __restrict__ gsrc,
                                              bf16* lds_dst) {
  const unsigned lds = (unsigned)(size_t)lds_dst;           // low 32 = LDS offset
  const unsigned long long ga = (unsigned long long)(size_t)gsrc;
  v4u g0;
  g0[0] = 1u;                                               // count=1, user mode
  g0[1] = lds;                                              // lds_addr
  g0[2] = (unsigned)ga;                                     // global_addr[31:0]
  g0[3] = ((unsigned)(ga >> 32) & 0x01FFFFFFu) | 0x80000000u; // addr[56:32]|type=2
  v8i g1;
  g1[0] = 0x00010000;            // workgroup_mask=0, data_size=1 (2 bytes)
  g1[1] = (int)(KPAD << 16);     // tensor_dim0[15:0] in [31:16]
  g1[2] = (int)(M_PTS << 16);    // tensor_dim0[31:16]=0 | tensor_dim1[15:0]
  g1[3] = (int)(KC << 16);       // tensor_dim1[31:16]=0 | tile_dim0
  g1[4] = 16;                    // tile_dim1=16, tile_dim2=0
  g1[5] = KPAD;                  // tensor_dim0_stride[31:0]
  g1[6] = 0;                     // stride hi / dim1_stride lo (unused, 2D tile)
  g1[7] = 0;
  v4i z4 = {0, 0, 0, 0};
  v8i z8 = {0, 0, 0, 0, 0, 0, 0, 0};
  __builtin_amdgcn_tensor_load_to_lds(g0, g1, z4, z4, z8, 0);
}

// 16-bit A/B fragment (16x32) from an LDS panel with row stride KC:
// lane l -> row M = l&15; half = l>>4; elements 0..7 hold k = 8*half+0..7,
// elements 8..15 hold k = 8*half+16..23 (relative to kloc).
__device__ __forceinline__ v16bf frag_lds(const bf16* __restrict__ p0,
                                          int kloc, int lane) {
  const int m = lane & 15;
  const int base = (lane >> 4) << 3;  // 0 or 8
  const bf16* p = p0 + (size_t)m * KC + kloc + base;
  v8bf lo = *(const v8bf*)(p);
  v8bf hi = *(const v8bf*)(p + 16);
  v16bf f;
#pragma unroll
  for (int e = 0; e < 8; ++e) { f[e] = lo[e]; f[e + 8] = hi[e]; }
  return f;
}

__global__ void __launch_bounds__(256) k_gram(const bf16* __restrict__ xbf,
                                              const float* __restrict__ rx,
                                              float* __restrict__ Dx) {
  // double-buffered panels: A = 32 rows, B = 64 rows, KC cols each
  __shared__ __align__(16) bf16 Abuf[2][32 * KC];   // 20 KB
  __shared__ __align__(16) bf16 Bbuf[2][64 * KC];   // 40 KB
  const int lane = threadIdx.x & 31;
  const int wave = threadIdx.x >> 5;
  const int bi = blockIdx.x >> 5;   // 64 macro-rows of 32
  const int bj = blockIdx.x & 31;   // 32 macro-cols of 64
  const int wi = wave >> 2;         // 0..1
  const int wj = wave & 3;          // 0..3
  const int row0 = bi * 32;
  const int col0 = bj * 64;

  // loader roles: waves 0,1 load A slices; waves 2..5 load B slices
  const bool loader = (wave < 6);
  const int lrow = (wave < 2) ? (row0 + wave * 16) : (col0 + (wave - 2) * 16);
  const int lofs = (wave < 2) ? (wave * 16 * KC) : ((wave - 2) * 16 * KC);

  if (loader) {
    bf16* dst = (wave < 2) ? &Abuf[0][lofs] : &Bbuf[0][lofs];
    tdm_load_tile(xbf + (size_t)lrow * KPAD, dst);
  }

  v8f acc = {};
  for (int c = 0; c < NCHUNK; ++c) {
    const int cur = c & 1;
    if (c + 1 < NCHUNK) {
      if (loader) {
        bf16* dst = (wave < 2) ? &Abuf[cur ^ 1][lofs] : &Bbuf[cur ^ 1][lofs];
        tdm_load_tile(xbf + (size_t)lrow * KPAD + (size_t)(c + 1) * KC, dst);
        __builtin_amdgcn_s_wait_tensorcnt(1);   // chunk c done (in-order per wave)
      }
    } else if (loader) {
      __builtin_amdgcn_s_wait_tensorcnt(0);
    }
    __syncthreads();                            // publish chunk c to all waves
    asm volatile("" ::: "memory");
    const bf16* pa = &Abuf[cur][wi * 16 * KC];
    const bf16* pb = &Bbuf[cur][wj * 16 * KC];
#pragma unroll
    for (int kloc = 0; kloc < KC; kloc += 32) {
      v16bf a = frag_lds(pa, kloc, lane);
      v16bf b = frag_lds(pb, kloc, lane);       // X X^T: same fragment layout
      acc = __builtin_amdgcn_wmma_f32_16x16x32_bf16(
          false, a, false, b, (short)0, acc, false, false);
    }
    __syncthreads();  // all reads of buf `cur` done before it is refilled (c+2)
  }

  const int j = col0 + wj * 16 + (lane & 15);
  const int half = lane >> 4;
  const float rj = rx[j];
#pragma unroll
  for (int v = 0; v < 8; ++v) {
    const int i = row0 + wi * 16 + v + (half << 3);
    float d = rx[i] + rj - 2.0f * acc[v];
    if (i == j) d = 0.0f;
    Dx[(size_t)i * M_PTS + j] = d;
  }
}

// ---- 3) Dy directly -------------------------------------------------------
__global__ void k_dy(const float* __restrict__ y, float* __restrict__ Dy) {
  const int i = blockIdx.x;
  float yi[DLAT];
#pragma unroll
  for (int a = 0; a < DLAT; ++a) yi[a] = y[(size_t)i * DLAT + a];
  for (int j = threadIdx.x; j < M_PTS; j += blockDim.x) {
    float s = 0.0f;
#pragma unroll
    for (int a = 0; a < DLAT; ++a) {
      float dv = yi[a] - y[(size_t)j * DLAT + a];
      s += dv * dv;
    }
    Dy[(size_t)i * M_PTS + j] = s;
  }
}

// ---- 4a) max of a nonneg buffer (uint-bit atomicMax trick) ----------------
__global__ void k_max(const float* __restrict__ buf, unsigned long long n,
                      unsigned* __restrict__ out) {
  float mx = 0.0f;
  for (unsigned long long idx = (unsigned long long)blockIdx.x * blockDim.x + threadIdx.x;
       idx < n; idx += (unsigned long long)gridDim.x * blockDim.x)
    mx = fmaxf(mx, buf[idx]);
#pragma unroll
  for (int o = 16; o > 0; o >>= 1) mx = fmaxf(mx, __shfl_down(mx, o, 32));
  __shared__ float sm[32];
  const int lane = threadIdx.x & 31, wv = threadIdx.x >> 5;
  if (lane == 0) sm[wv] = mx;
  __syncthreads();
  if (threadIdx.x == 0) {
    float t = sm[0];
    for (int k = 1; k < (int)(blockDim.x >> 5); ++k) t = fmaxf(t, sm[k]);
    atomicMax(out, __float_as_uint(fmaxf(t, 0.0f)));
  }
}

// ---- 4b) per-component min/max of y ---------------------------------------
__global__ void k_ymm(const float* __restrict__ y, float* __restrict__ ymin,
                      float* __restrict__ ymax) {
  __shared__ float smn[256], smx[256];
  const int a = blockIdx.x;
  float mn = 3.4e38f, mx = -3.4e38f;
  for (int i = threadIdx.x; i < M_PTS; i += blockDim.x) {
    float v = y[(size_t)i * DLAT + a];
    mn = fminf(mn, v);
    mx = fmaxf(mx, v);
  }
  smn[threadIdx.x] = mn; smx[threadIdx.x] = mx;
  __syncthreads();
  for (int s = (int)blockDim.x >> 1; s > 0; s >>= 1) {
    if ((int)threadIdx.x < s) {
      smn[threadIdx.x] = fminf(smn[threadIdx.x], smn[threadIdx.x + s]);
      smx[threadIdx.x] = fmaxf(smx[threadIdx.x], smx[threadIdx.x + s]);
    }
    __syncthreads();
  }
  if (threadIdx.x == 0) { ymin[a] = smn[0]; ymax[a] = smx[0]; }
}

// ---- 4c) init selection state ---------------------------------------------
__global__ void k_selinit(Sel* __restrict__ sels, const unsigned* __restrict__ maxb,
                          const float* __restrict__ ymin, const float* __restrict__ ymax) {
  const int s = threadIdx.x;
  if (s >= NSEL) return;
  float hi;
  if (s < 2) {
    hi = __uint_as_float(maxb[s]);
  } else {
    float r = ymax[s - 2] - ymin[s - 2];
    hi = r * r;
  }
  hi = hi * 1.000001f + 1e-30f;
  sels[s].lo = 0.0f;
  sels[s].hi = hi;
  sels[s].rank = RANKSEL;
}

// ---- 4d) histogram of a distance buffer within [lo,hi) --------------------
__global__ void k_hist_buf(const float* __restrict__ buf, const Sel* __restrict__ selp,
                           unsigned* __restrict__ hist) {
  __shared__ unsigned sh[NBF];
  for (int b = threadIdx.x; b < NBF; b += blockDim.x) sh[b] = 0u;
  __syncthreads();
  const float lo = selp->lo, hi = selp->hi;
  const float scale = (float)NBF / (hi - lo);
  const unsigned long long n = (unsigned long long)M_PTS * M_PTS;
  for (unsigned long long idx = (unsigned long long)blockIdx.x * blockDim.x + threadIdx.x;
       idx < n; idx += (unsigned long long)gridDim.x * blockDim.x) {
    float v = buf[idx];
    if (v >= lo && v < hi) {
      int b = (int)((v - lo) * scale);
      b = min(b, NBF - 1);
      atomicAdd(&sh[b], 1u);
    }
  }
  __syncthreads();
  for (int b = threadIdx.x; b < NBF; b += blockDim.x)
    if (sh[b]) atomicAdd(&hist[b], sh[b]);
}

// ---- 4e) histograms of the 10 on-the-fly component distances --------------
__global__ void k_hist_y(const float* __restrict__ y, const Sel* __restrict__ sels,
                         unsigned* __restrict__ hist) {  // hist stride HSTRIDE per comp
  __shared__ unsigned sh[DLAT * NBC];  // 40 KB
  for (int b = threadIdx.x; b < DLAT * NBC; b += blockDim.x) sh[b] = 0u;
  __syncthreads();
  float lo[DLAT], hi[DLAT], sc[DLAT];
#pragma unroll
  for (int a = 0; a < DLAT; ++a) {
    lo[a] = sels[a].lo;
    hi[a] = sels[a].hi;
    sc[a] = (float)NBC / (hi[a] - lo[a]);
  }
  for (int i = blockIdx.x; i < M_PTS; i += gridDim.x) {
    float yi[DLAT];
#pragma unroll
    for (int a = 0; a < DLAT; ++a) yi[a] = y[(size_t)i * DLAT + a];
    for (int j = threadIdx.x; j < M_PTS; j += blockDim.x) {
#pragma unroll
      for (int a = 0; a < DLAT; ++a) {
        float dv = yi[a] - y[(size_t)j * DLAT + a];
        float v = dv * dv;
        if (v >= lo[a] && v < hi[a]) {
          int b = (int)((v - lo[a]) * sc[a]);
          b = min(b, NBC - 1);
          atomicAdd(&sh[a * NBC + b], 1u);
        }
      }
    }
  }
  __syncthreads();
  for (int t = threadIdx.x; t < DLAT * NBC; t += blockDim.x) {
    int a = t / NBC, b = t % NBC;
    if (sh[t]) atomicAdd(&hist[(size_t)a * HSTRIDE + b], sh[t]);
  }
}

// ---- 4f) scan histogram, refine range or emit sigma -----------------------
__global__ void k_scan(const unsigned* __restrict__ hist, Sel* __restrict__ sels,
                       float* __restrict__ inv2s2, int final_level) {
  if (threadIdx.x != 0) return;
  const int s = blockIdx.x;
  const int nb = (s < 2) ? NBF : NBC;
  const unsigned* h = hist + (size_t)s * HSTRIDE;
  Sel sel = sels[s];
  const float w = (sel.hi - sel.lo) / (float)nb;
  unsigned long long cum = 0, before = 0;
  int bs = nb - 1;
  bool found = false;
  for (int b = 0; b < nb && !found; ++b) {
    if (cum + h[b] >= sel.rank) { bs = b; before = cum; found = true; }
    else cum += h[b];
  }
  if (!found) {
    bs = nb - 1;
    before = (cum >= (unsigned long long)h[nb - 1]) ? cum - h[nb - 1] : 0ULL;
  }
  if (final_level) {
    float med = sel.lo + ((float)bs + 0.5f) * w;
    float sig = fmaxf(med, 0.01f);
    inv2s2[s] = 1.0f / (2.0f * sig * sig);
  } else {
    sels[s].lo = sel.lo + (float)bs * w;
    sels[s].hi = sel.lo + (float)(bs + 1) * w;
    sels[s].rank = sel.rank - before;
  }
}

// ---- 5a) mi_io reductions -------------------------------------------------
__global__ void k_mired(const float* __restrict__ Dx, const float* __restrict__ Dy,
                        const float* __restrict__ inv2s2, float* __restrict__ rowSx,
                        float* __restrict__ rowSy, float* __restrict__ scal) {
  __shared__ float sm[32];
  const int i = blockIdx.x;
  const float i2x = inv2s2[0], i2y = inv2s2[1];
  const float* dxr = Dx + (size_t)i * M_PTS;
  const float* dyr = Dy + (size_t)i * M_PTS;
  float sx = 0, sy = 0, sxy = 0, sxx = 0, syy = 0;
  for (int j = threadIdx.x; j < M_PTS; j += blockDim.x) {
    float kx = __expf(-dxr[j] * i2x);
    float ky = __expf(-dyr[j] * i2y);
    sx += kx; sy += ky; sxy += kx * ky; sxx += kx * kx; syy += ky * ky;
  }
  sx  = block_reduce(sx, sm);
  sy  = block_reduce(sy, sm);
  sxy = block_reduce(sxy, sm);
  sxx = block_reduce(sxx, sm);
  syy = block_reduce(syy, sm);
  if (threadIdx.x == 0) {
    rowSx[i] = sx;
    rowSy[i] = sy;
    atomicAdd(&scal[0], sxy);
    atomicAdd(&scal[1], sxx);
    atomicAdd(&scal[2], syy);
  }
}

// ---- 5b) component-pair reductions ----------------------------------------
__global__ void k_compred(const float* __restrict__ y, const float* __restrict__ i2s,
                          float* __restrict__ rowSc, float* __restrict__ gab) {
  __shared__ float sm[32];
  const int i = blockIdx.x;
  float yi[DLAT], i2[DLAT], rs[DLAT], g[NPAIR];
#pragma unroll
  for (int a = 0; a < DLAT; ++a) {
    yi[a] = y[(size_t)i * DLAT + a];
    i2[a] = i2s[a];
    rs[a] = 0.0f;
  }
#pragma unroll
  for (int t = 0; t < NPAIR; ++t) g[t] = 0.0f;
  for (int j = threadIdx.x; j < M_PTS; j += blockDim.x) {
    float k[DLAT];
#pragma unroll
    for (int a = 0; a < DLAT; ++a) {
      float dv = yi[a] - y[(size_t)j * DLAT + a];
      k[a] = __expf(-dv * dv * i2[a]);
      rs[a] += k[a];
    }
    int idx = 0;
#pragma unroll
    for (int a = 0; a < DLAT; ++a)
#pragma unroll
      for (int b = a; b < DLAT; ++b) { g[idx] += k[a] * k[b]; ++idx; }
  }
#pragma unroll
  for (int a = 0; a < DLAT; ++a) {
    float t = block_reduce(rs[a], sm);
    if (threadIdx.x == 0) rowSc[(size_t)a * M_PTS + i] = t;
  }
  for (int t = 0; t < NPAIR; ++t) {
    float gv = block_reduce(g[t], sm);
    if (threadIdx.x == 0) atomicAdd(&gab[t], gv);
  }
}

// ---- 5c) finalize ---------------------------------------------------------
__global__ void k_final(const float* __restrict__ rowSx, const float* __restrict__ rowSy,
                        const float* __restrict__ rowSc, const float* __restrict__ scal,
                        const float* __restrict__ gab, float* __restrict__ out) {
  __shared__ float sm[32];
  float vals[80];
#pragma unroll
  for (int t = 0; t < 80; ++t) vals[t] = 0.0f;
  // vals[0..4] = sumSx, sumSy, dot(Sx,Sy), dot(Sx,Sx), dot(Sy,Sy)
  // vals[5..14] = sum(S_a); vals[15..79] = dot(S_a,S_b) for a<=b
  for (int i = threadIdx.x; i < M_PTS; i += blockDim.x) {
    float sx = rowSx[i], sy = rowSy[i];
    vals[0] += sx; vals[1] += sy; vals[2] += sx * sy;
    vals[3] += sx * sx; vals[4] += sy * sy;
    float sc[DLAT];
#pragma unroll
    for (int a = 0; a < DLAT; ++a) {
      sc[a] = rowSc[(size_t)a * M_PTS + i];
      vals[5 + a] += sc[a];
    }
    int idx = 0;
#pragma unroll
    for (int a = 0; a < DLAT; ++a)
#pragma unroll
      for (int b = a; b < DLAT; ++b) { vals[15 + idx] += sc[a] * sc[b]; ++idx; }
  }
  float red[80];
  for (int t = 0; t < 80; ++t) red[t] = block_reduce(vals[t], sm);
  if (threadIdx.x == 0) {
    const float m = (float)M_PTS;
    const float m2 = m * m;
    const float Pxy = scal[0] - (2.0f / m) * red[2] + (red[0] * red[1]) / m2;
    const float Pxx = scal[1] - (2.0f / m) * red[3] + (red[0] * red[0]) / m2;
    const float Pyy = scal[2] - (2.0f / m) * red[4] + (red[1] * red[1]) / m2;
    const float mi_io = Pxy / sqrtf(Pxx * Pyy);
    float P[NPAIR], diag[DLAT];
    int idx = 0;
    for (int a = 0; a < DLAT; ++a)
      for (int b = a; b < DLAT; ++b) {
        float pab = gab[idx] - (2.0f / m) * red[15 + idx]
                  + (red[5 + a] * red[5 + b]) / m2;
        P[idx] = pab;
        if (a == b) diag[a] = pab;
        ++idx;
      }
    float tot = 0.0f;
    idx = 0;
    for (int a = 0; a < DLAT; ++a)
      for (int b = a; b < DLAT; ++b) {
        if (b > a) tot += P[idx] / sqrtf(diag[a] * diag[b]);
        ++idx;
      }
    out[0] = -mi_io + tot;
  }
}

// ---------------------------------------------------------------------------
extern "C" void kernel_launch(void* const* d_in, const int* in_sizes, int n_in,
                              void* d_out, int out_size, void* d_ws, size_t ws_size,
                              hipStream_t stream) {
  (void)in_sizes; (void)n_in; (void)out_size; (void)ws_size;
  const float* x = (const float*)d_in[0];
  const float* y = (const float*)d_in[1];
  float* out = (float*)d_out;
  char* ws = (char*)d_ws;

  bf16*     xbf   = (bf16*)(ws + OFF_XBF);
  float*    rx    = (float*)(ws + OFF_RX);
  float*    Dx    = (float*)(ws + OFF_DX);
  float*    Dy    = (float*)(ws + OFF_DY);
  unsigned* hist  = (unsigned*)(ws + OFF_HIST);
  Sel*      sels  = (Sel*)(ws + OFF_SEL);
  float*    inv2  = (float*)(ws + OFF_INV);
  unsigned* maxb  = (unsigned*)(ws + OFF_MAXB);
  float*    ymin  = (float*)(ws + OFF_YMM);
  float*    ymax  = (float*)(ws + OFF_YMM) + DLAT;
  float*    scal  = (float*)(ws + OFF_SCAL);
  float*    gab   = (float*)(ws + OFF_GAB);
  float*    rowSx = (float*)(ws + OFF_RSX);
  float*    rowSy = (float*)(ws + OFF_RSY);
  float*    rowSc = (float*)(ws + OFF_RSC);

  const unsigned long long n2 = (unsigned long long)M_PTS * M_PTS;

  // stage 1-3
  k_prep_x<<<M_PTS, 256, 0, stream>>>(x, xbf, rx);
  k_gram<<<2048, 256, 0, stream>>>(xbf, rx, Dx);
  k_dy<<<M_PTS, 256, 0, stream>>>(y, Dy);

  // stage 4: sigmas
  (void)hipMemsetAsync(maxb, 0, 2 * sizeof(unsigned), stream);
  k_max<<<512, 256, 0, stream>>>(Dx, n2, maxb + 0);
  k_max<<<512, 256, 0, stream>>>(Dy, n2, maxb + 1);
  k_ymm<<<DLAT, 256, 0, stream>>>(y, ymin, ymax);
  k_selinit<<<1, 32, 0, stream>>>(sels, maxb, ymin, ymax);
  for (int level = 0; level < 2; ++level) {
    (void)hipMemsetAsync(hist, 0, (size_t)NSEL * HSTRIDE * sizeof(unsigned), stream);
    k_hist_buf<<<512, 256, 0, stream>>>(Dx, sels + 0, hist + 0 * HSTRIDE);
    k_hist_buf<<<512, 256, 0, stream>>>(Dy, sels + 1, hist + 1 * HSTRIDE);
    k_hist_y<<<512, 256, 0, stream>>>(y, sels + 2, hist + 2 * HSTRIDE);
    k_scan<<<NSEL, 32, 0, stream>>>(hist, sels, inv2, (level == 1) ? 1 : 0);
  }

  // stage 5: HSIC reductions
  (void)hipMemsetAsync(scal, 0, 16 * sizeof(float), stream);
  (void)hipMemsetAsync(gab, 0, NPAIR * sizeof(float), stream);
  k_mired<<<M_PTS, 256, 0, stream>>>(Dx, Dy, inv2, rowSx, rowSy, scal);
  k_compred<<<M_PTS, 256, 0, stream>>>(y, inv2 + 2, rowSc, gab);
  k_final<<<1, 256, 0, stream>>>(rowSx, rowSy, rowSc, scal, gab, out);
}
